// Multi_Head_Attention_86002425135115
// MI455X (gfx1250) — compile-verified, matrix operand order checked by reading, not run on hardware
//
#include <hip/hip_runtime.h>

// MI455X / gfx1250 multi-head attention: bf16 WMMA + flash softmax + TDM.
// B=4, S=2048, DM=1024, H=16, HD=64.
// Workspace (u16 elements): q[8M] k[8M] vT[8M] attn[8M] = 64 MB.
// Score scale 1/sqrt(64) is folded into the q projection epilogue.

typedef __attribute__((ext_vector_type(16))) __bf16 v16bf;
typedef __attribute__((ext_vector_type(8)))  float  v8f;
typedef __attribute__((ext_vector_type(4)))  float  f32x4;
typedef unsigned short u16;
typedef __attribute__((ext_vector_type(8)))  u16    u16x8;
typedef __attribute__((ext_vector_type(4)))  unsigned int u32x4;
typedef __attribute__((ext_vector_type(8)))  int    i32x8;
typedef __attribute__((ext_vector_type(4)))  int    i32x4;

union Frag {
  v16bf bf;
  u16x8 h[2];
};

__device__ __forceinline__ u16 f2bf(float f) {
  unsigned u = __builtin_bit_cast(unsigned, f);
  u += 0x7FFFu + ((u >> 16) & 1u);  // round-to-nearest-even
  return (u16)(u >> 16);
}
__device__ __forceinline__ unsigned f2bf_pk(float lo, float hi) {
#if __has_builtin(__builtin_amdgcn_cvt_pk_bf16_f32)
  auto r = __builtin_amdgcn_cvt_pk_bf16_f32(lo, hi);  // v_cvt_pk_bf16_f32
  return __builtin_bit_cast(unsigned, r);
#else
  return (unsigned)f2bf(lo) | ((unsigned)f2bf(hi) << 16);
#endif
}

__device__ __forceinline__ v8f wmma_bf16(const Frag& a, const Frag& b, v8f c) {
  return __builtin_amdgcn_wmma_f32_16x16x32_bf16(
      false, a.bf, false, b.bf, (short)0, c, false, false);
}

__device__ __forceinline__ float red_max16(float v) {
  v = fmaxf(v, __shfl_xor(v, 1));
  v = fmaxf(v, __shfl_xor(v, 2));
  v = fmaxf(v, __shfl_xor(v, 4));
  v = fmaxf(v, __shfl_xor(v, 8));
  return v;
}
__device__ __forceinline__ float red_sum16(float v) {
  v += __shfl_xor(v, 1);
  v += __shfl_xor(v, 2);
  v += __shfl_xor(v, 4);
  v += __shfl_xor(v, 8);
  return v;
}

// ---------------------------------------------------------------------------
// Kernel 1: fused QKV projections. grid = (32, 16, 3), block = 256 (8 waves).
// Workgroup tile 256M x 64N; each wave 32M x 64N (8 WMMAs / K-step).
// z=0: q*0.125 -> q_ws [B,H,S,HD]; z=1: k -> k_ws [B,H,S,HD];
// z=2: v -> vT_ws [B,H,HD,S]
// ---------------------------------------------------------------------------
__global__ __launch_bounds__(256) void proj_qkv_kernel(
    const float* __restrict__ Qin, const float* __restrict__ Kin,
    const float* __restrict__ Vin,
    const float* __restrict__ Wq, const float* __restrict__ bq,
    const float* __restrict__ Wk, const float* __restrict__ bk,
    const float* __restrict__ Wv, const float* __restrict__ bv,
    u16* __restrict__ q_ws, u16* __restrict__ k_ws, u16* __restrict__ vT_ws) {
  __shared__ __align__(16) u16 lA[256 * 32];  // 16 KB A tile (bf16)
  __shared__ __align__(16) u16 lBt[64 * 32];  //  4 KB B tile transposed [n][k]

  const int t = threadIdx.x;
  const int z = blockIdx.z;
  const float* X    = (z == 0) ? Qin : ((z == 1) ? Kin : Vin);
  const float* W    = (z == 0) ? Wq  : ((z == 1) ? Wk  : Wv);
  const float* bias = (z == 0) ? bq  : ((z == 1) ? bk  : bv);
  u16* dstQK        = (z == 0) ? q_ws : k_ws;
  const float oscale = (z == 0) ? 0.125f : 1.0f;  // fold 1/sqrt(HD) into q

  const int rBase = blockIdx.x * 256;
  const int nBase = blockIdx.y * 64;
  const int w  = t >> 5;
  const int L  = t & 31;
  const int lr = L & 15;
  const int hk = (L >> 4) * 8;

  const v8f vzero = {};
  v8f acc[8] = {vzero, vzero, vzero, vzero, vzero, vzero, vzero, vzero};

  const int bk_ = t >> 3;        // 0..31 (k row of B tile)
  const int bn  = (t & 7) * 8;   // 0..56 (n chunk)

  for (int kb = 0; kb < 1024; kb += 32) {
    // stage A: one 32-float row per thread, fp32 -> packed bf16 LDS
    {
      const float* src = X + (size_t)(rBase + t) * 1024 + kb;
      float xs[32];
      #pragma unroll
      for (int i = 0; i < 8; ++i)
        *(f32x4*)(xs + 4 * i) = *(const f32x4*)(src + 4 * i);
      u16* dst = &lA[t * 32];
      #pragma unroll
      for (int i = 0; i < 16; ++i)
        *(unsigned*)(dst + 2 * i) = f2bf_pk(xs[2 * i], xs[2 * i + 1]);
    }
    // stage B transposed: W[kb+k][nBase+n] -> lBt[n][k]
    {
      const float* src = W + (size_t)(kb + bk_) * 1024 + nBase + bn;
      float xs[8];
      *(f32x4*)(xs + 0) = *(const f32x4*)(src + 0);
      *(f32x4*)(xs + 4) = *(const f32x4*)(src + 4);
      #pragma unroll
      for (int i = 0; i < 8; ++i) lBt[(bn + i) * 32 + bk_] = f2bf(xs[i]);
    }
    __syncthreads();

    Frag fa0, fa1;
    const u16* ap0 = &lA[(w * 32 + lr) * 32];
    const u16* ap1 = &lA[(w * 32 + 16 + lr) * 32];
    fa0.h[0] = *(const u16x8*)(ap0 + hk);
    fa0.h[1] = *(const u16x8*)(ap0 + 16 + hk);
    fa1.h[0] = *(const u16x8*)(ap1 + hk);
    fa1.h[1] = *(const u16x8*)(ap1 + 16 + hk);
    #pragma unroll
    for (int ns = 0; ns < 4; ++ns) {
      Frag fb;
      const u16* bp = &lBt[(ns * 16 + lr) * 32];
      fb.h[0] = *(const u16x8*)(bp + hk);
      fb.h[1] = *(const u16x8*)(bp + 16 + hk);
      acc[ns]     = wmma_bf16(fa0, fb, acc[ns]);
      acc[4 + ns] = wmma_bf16(fa1, fb, acc[4 + ns]);
    }
    __syncthreads();
  }

  // epilogue: bias (+q scale) + layout scatter; bI uniform per block
  const int bI    = rBase >> 11;
  const int sBase = (rBase & 2047) + w * 32 + ((L >> 4) << 3);
  #pragma unroll
  for (int ns = 0; ns < 4; ++ns) {
    const int col = nBase + ns * 16 + lr;
    const float bb = bias[col];
    const int hh = col >> 6, dd = col & 63;
    const size_t qkBase = ((size_t)(bI * 16 + hh) * 2048) * 64 + dd;
    const size_t vBase  = ((size_t)(bI * 16 + hh) * 64 + dd) * 2048;
    #pragma unroll
    for (int ms = 0; ms < 2; ++ms) {
      #pragma unroll
      for (int j = 0; j < 8; ++j) {
        const int sI = sBase + ms * 16 + j;
        const u16 val = f2bf((acc[ms * 4 + ns][j] + bb) * oscale);
        if (z == 2) vT_ws[vBase + sI] = val;
        else        dstQK[qkBase + (size_t)sI * 64] = val;
      }
    }
  }
}

// ---------------------------------------------------------------------------
// Kernel 2: flash attention. grid = (S/128, H, B) = (16,16,4), block = 256.
// Each wave owns 16 q-rows; 64 keys per iteration (16 WMMAs / iteration).
// ---------------------------------------------------------------------------
__global__ __launch_bounds__(256) void flash_attn_kernel(
    const u16* __restrict__ q_ws, const u16* __restrict__ k_ws,
    const u16* __restrict__ vT_ws, const int* __restrict__ mask,
    u16* __restrict__ attn_ws) {
  __shared__ __align__(16) u16 lP[8 * 16 * 64];  // per-wave P staging, 16 KB

  const int t = threadIdx.x;
  const int w  = t >> 5;
  const int L  = t & 31;
  const int lr = L & 15;
  const int hk = (L >> 4) * 8;
  const int bI = blockIdx.z;
  const int h  = blockIdx.y;
  const int qBase = blockIdx.x * 128;
  const int bh = bI * 16 + h;

  // Q fragments (HD=64 => 2 A fragments), resident for whole kernel.
  // q was pre-scaled by 1/sqrt(HD) in the projection.
  const u16* qp = q_ws + ((size_t)bh * 2048 + qBase + w * 16 + lr) * 64;
  Frag fq[2];
  fq[0].h[0] = *(const u16x8*)(qp + 0  + hk);
  fq[0].h[1] = *(const u16x8*)(qp + 16 + hk);
  fq[1].h[0] = *(const u16x8*)(qp + 32 + hk);
  fq[1].h[1] = *(const u16x8*)(qp + 48 + hk);

  const u16* kBasePtr = k_ws + (size_t)bh * 2048 * 64;
  const u16* vBasePtr = vT_ws + (size_t)bh * 64 * 2048;

  const v8f vzero = {};
  v8f o[4] = {vzero, vzero, vzero, vzero};
  float m_r[8], l_r[8];
  #pragma unroll
  for (int j = 0; j < 8; ++j) { m_r[j] = -3.0e38f; l_r[j] = 0.0f; }

  for (int kb = 0; kb < 2048; kb += 64) {
    // prefetch next 64-key K/V block into cache (global_prefetch_b8)
    if (kb + 64 < 2048) {
      __builtin_prefetch(kBasePtr + (size_t)(kb + 64 + lr) * 64, 0, 3);
      __builtin_prefetch(vBasePtr + (size_t)lr * 2048 + kb + 64, 0, 3);
    }

    // ---- S = Q K^T : four 16-key tiles ----
    v8f s[4] = {vzero, vzero, vzero, vzero};
    #pragma unroll
    for (int f = 0; f < 2; ++f) {
      #pragma unroll
      for (int tl = 0; tl < 4; ++tl) {
        const u16* kp = kBasePtr + (size_t)(kb + tl * 16 + lr) * 64 + f * 32;
        Frag fk;
        fk.h[0] = *(const u16x8*)(kp + hk);
        fk.h[1] = *(const u16x8*)(kp + 16 + hk);
        s[tl] = wmma_bf16(fq[f], fk, s[tl]);
      }
    }
    float madd[4];
    #pragma unroll
    for (int tl = 0; tl < 4; ++tl)
      madd[tl] = mask[bI * 2048 + kb + tl * 16 + lr] ? 0.0f : -3.0e38f;

    // ---- online softmax over 64 keys ----
    #pragma unroll
    for (int j = 0; j < 8; ++j) {
      float a0 = s[0][j] + madd[0];
      float a1 = s[1][j] + madd[1];
      float a2 = s[2][j] + madd[2];
      float a3 = s[3][j] + madd[3];
      float rm = red_max16(fmaxf(fmaxf(a0, a1), fmaxf(a2, a3)));
      float mn = fmaxf(m_r[j], rm);
      float corr = __expf(m_r[j] - mn);
      m_r[j] = mn;
      float p0 = __expf(a0 - mn), p1 = __expf(a1 - mn);
      float p2 = __expf(a2 - mn), p3 = __expf(a3 - mn);
      float rs = red_sum16((p0 + p1) + (p2 + p3));
      l_r[j] = l_r[j] * corr + rs;
      o[0][j] *= corr; o[1][j] *= corr; o[2][j] *= corr; o[3][j] *= corr;
      const int R = j + ((L >> 4) << 3);
      u16* pr = &lP[w * 1024 + R * 64 + lr];
      pr[0]  = f2bf(p0);
      pr[16] = f2bf(p1);
      pr[32] = f2bf(p2);
      pr[48] = f2bf(p3);
    }
    asm volatile("s_wait_dscnt 0" ::: "memory");  // P stores -> fragment loads

    // ---- O += P V : 2 P-fragments x 4 hd-subtiles ----
    Frag fp[2];
    const u16* pp = &lP[w * 1024 + lr * 64];
    #pragma unroll
    for (int kf = 0; kf < 2; ++kf) {
      fp[kf].h[0] = *(const u16x8*)(pp + kf * 32 + hk);
      fp[kf].h[1] = *(const u16x8*)(pp + kf * 32 + 16 + hk);
    }
    #pragma unroll
    for (int hd = 0; hd < 4; ++hd) {
      const u16* vp = vBasePtr + (size_t)(hd * 16 + lr) * 2048 + kb;
      #pragma unroll
      for (int kf = 0; kf < 2; ++kf) {
        Frag fv;
        fv.h[0] = *(const u16x8*)(vp + kf * 32 + hk);
        fv.h[1] = *(const u16x8*)(vp + kf * 32 + 16 + hk);
        o[hd] = wmma_bf16(fp[kf], fv, o[hd]);
      }
    }
  }

  // normalize and write [B,S,DM] bf16
  #pragma unroll
  for (int j = 0; j < 8; ++j) {
    const float inv = 1.0f / l_r[j];
    const int row = qBase + w * 16 + j + ((L >> 4) << 3);
    u16* dst = attn_ws + ((size_t)bI * 2048 + row) * 1024 + h * 64;
    #pragma unroll
    for (int hd = 0; hd < 4; ++hd)
      dst[hd * 16 + lr] = f2bf(o[hd][j] * inv);
  }
}

// ---------------------------------------------------------------------------
// Kernel 3: out = attn @ Wo + bo (fp32). grid = (32, 16), block = 256.
// A tile (already bf16) is DMA'd to LDS by the Tensor Data Mover (TDM).
// ---------------------------------------------------------------------------
__global__ __launch_bounds__(256) void proj_out_kernel(
    const u16* __restrict__ attn, const float* __restrict__ Wo,
    const float* __restrict__ bo, float* __restrict__ out) {
  __shared__ __align__(16) u16 lA[256 * 32];  // 16 KB
  __shared__ __align__(16) u16 lBt[64 * 32];  //  4 KB

  const int t = threadIdx.x;
  const int rBase = blockIdx.x * 256;
  const int nBase = blockIdx.y * 64;
  const int w  = t >> 5;
  const int L  = t & 31;
  const int lr = L & 15;
  const int hk = (L >> 4) * 8;

  const v8f vzero = {};
  v8f acc[8] = {vzero, vzero, vzero, vzero, vzero, vzero, vzero, vzero};

  const int bk_ = t >> 3;
  const int bn  = (t & 7) * 8;

  for (int kb = 0; kb < 1024; kb += 32) {
    // ---- TDM: async 256x32 bf16 A tile, global -> LDS (wave 0 issues) ----
    if (w == 0) {
      const unsigned long long ga =
          (unsigned long long)(const void*)(attn + (size_t)rBase * 1024 + kb);
      const unsigned lds = (unsigned)(unsigned long long)(const void*)&lA[0];
      u32x4 g0;
      g0[0] = 1u;                                 // count=1 (valid user D#)
      g0[1] = lds;                                // lds_addr
      g0[2] = (unsigned)ga;                       // global_addr[31:0]
      g0[3] = (unsigned)((ga >> 32) & 0x01FFFFFFull) | (2u << 30);  // type=2
      i32x8 g1;
      g1[0] = 1 << 16;                            // data_size=1 (2 bytes)
      g1[1] = (int)((1024u & 0xFFFFu) << 16);     // tensor_dim0 lo (=1024)
      g1[2] = (int)((8192u & 0xFFFFu) << 16);     // dim0 hi(0) | tensor_dim1 lo
      g1[3] = (int)(32u << 16);                   // dim1 hi(0) | tile_dim0=32
      g1[4] = (int)256;                           // tile_dim1=256, tile_dim2=0
      g1[5] = (int)1024;                          // tensor_dim0_stride lo
      g1[6] = 0;                                  // stride hi | dim1_stride lo
      g1[7] = 0;
      const i32x4 z4 = {0, 0, 0, 0};
#if defined(__clang_major__) && (__clang_major__ >= 23)
      const i32x8 z8 = {0, 0, 0, 0, 0, 0, 0, 0};
      __builtin_amdgcn_tensor_load_to_lds(g0, g1, z4, z4, z8, 0);
#else
      __builtin_amdgcn_tensor_load_to_lds(g0, g1, z4, z4, 0);
#endif
      __builtin_amdgcn_s_wait_tensorcnt(0);
    }
    // stage B transposed (fp32 -> bf16)
    {
      const float* src = Wo + (size_t)(kb + bk_) * 1024 + nBase + bn;
      float xs[8];
      *(f32x4*)(xs + 0) = *(const f32x4*)(src + 0);
      *(f32x4*)(xs + 4) = *(const f32x4*)(src + 4);
      #pragma unroll
      for (int i = 0; i < 8; ++i) lBt[(bn + i) * 32 + bk_] = f2bf(xs[i]);
    }
    __syncthreads();

    Frag fa0, fa1;
    const u16* ap0 = &lA[(w * 32 + lr) * 32];
    const u16* ap1 = &lA[(w * 32 + 16 + lr) * 32];
    fa0.h[0] = *(const u16x8*)(ap0 + hk);
    fa0.h[1] = *(const u16x8*)(ap0 + 16 + hk);
    fa1.h[0] = *(const u16x8*)(ap1 + hk);
    fa1.h[1] = *(const u16x8*)(ap1 + 16 + hk);
    #pragma unroll
    for (int ns = 0; ns < 4; ++ns) {
      Frag fb;
      const u16* bp = &lBt[(ns * 16 + lr) * 32];
      fb.h[0] = *(const u16x8*)(bp + hk);
      fb.h[1] = *(const u16x8*)(bp + 16 + hk);
      acc[ns]     = wmma_bf16(fa0, fb, acc[ns]);
      acc[4 + ns] = wmma_bf16(fa1, fb, acc[4 + ns]);
    }
    __syncthreads();
  }

  #pragma unroll
  for (int ns = 0; ns < 4; ++ns) {
    const int col = nBase + ns * 16 + lr;
    const float bb = bo[col];
    #pragma unroll
    for (int ms = 0; ms < 2; ++ms) {
      #pragma unroll
      for (int j = 0; j < 8; ++j) {
        const int row = rBase + w * 32 + ms * 16 + j + ((L >> 4) << 3);
        out[(size_t)row * 1024 + col] = acc[ms * 4 + ns][j] + bb;
      }
    }
  }
}

// ---------------------------------------------------------------------------
extern "C" void kernel_launch(void* const* d_in, const int* in_sizes, int n_in,
                              void* d_out, int out_size, void* d_ws,
                              size_t ws_size, hipStream_t stream) {
  const float* Qin  = (const float*)d_in[0];
  const float* Kin  = (const float*)d_in[1];
  const float* Vin  = (const float*)d_in[2];
  const int*   mask = (const int*)d_in[3];
  const float* Wq = (const float*)d_in[4];
  const float* bq = (const float*)d_in[5];
  const float* Wk = (const float*)d_in[6];
  const float* bk = (const float*)d_in[7];
  const float* Wv = (const float*)d_in[8];
  const float* bv = (const float*)d_in[9];
  const float* Wo = (const float*)d_in[10];
  const float* bo = (const float*)d_in[11];

  const size_t N = (size_t)4 * 16 * 2048 * 64;  // 8,388,608 elems per buffer
  u16* q_ws    = (u16*)d_ws;
  u16* k_ws    = q_ws + N;
  u16* vT_ws   = q_ws + 2 * N;
  u16* attn_ws = q_ws + 3 * N;  // 64 MB total workspace

  proj_qkv_kernel<<<dim3(32, 16, 3), 256, 0, stream>>>(
      Qin, Kin, Vin, Wq, bq, Wk, bk, Wv, bv, q_ws, k_ws, vT_ws);
  flash_attn_kernel<<<dim3(16, 16, 4), 256, 0, stream>>>(
      q_ws, k_ws, vT_ws, mask, attn_ws);
  proj_out_kernel<<<dim3(32, 16), 256, 0, stream>>>(
      attn_ws, Wo, bo, (float*)d_out);
}
